// LSTMModel_42820823941592
// MI455X (gfx1250) — matrix operand order repaired and checked by reading, not compiled, not run on hardware
//
#include <hip/hip_runtime.h>
#include <hip/hip_bf16.h>

typedef _Float16 v16h __attribute__((ext_vector_type(16)));
typedef _Float16 v8h  __attribute__((ext_vector_type(8)));
typedef float    v8f  __attribute__((ext_vector_type(8)));

constexpr int Tc = 512;      // timesteps
constexpr int Bc = 4096;     // batch
constexpr int Hc = 50;       // hidden
constexpr int RS = 72;       // LDS row stride in halves (144B = 36 dwords -> conflict-free lane walk)
constexpr int WROWS = 272;   // 16 gate N-tiles * 16 + 1 output N-tile * 16

static __device__ __forceinline__ float ftanh(float x) {
#if __has_builtin(__builtin_amdgcn_tanhf)
  return __builtin_amdgcn_tanhf(x);          // v_tanh_f32 on gfx1250
#else
  float e = __expf(-2.0f * x);               // v_exp_f32 path
  return (1.0f - e) / (1.0f + e);
#endif
}
static __device__ __forceinline__ float fsig(float x) {
  return 0.5f * ftanh(0.5f * x) + 0.5f;
}

// A operand (16x32 f16): lane holds row (lane&15); halves = K {kh..kh+7, kh+16..kh+23},
// kh = (lane>=16 ? 8 : 0) + 32*kc. Two contiguous 16B LDS loads.
static __device__ __forceinline__ v16h ldsA(const _Float16* hb, int kc, int lane) {
  const int row = lane & 15;
  const int kh  = ((lane >> 4) << 3) + (kc << 5);
  const v8h lo = *(const v8h*)(hb + row * RS + kh);
  const v8h hi = *(const v8h*)(hb + row * RS + kh + 16);
  return __builtin_shufflevector(lo, hi, 0,1,2,3,4,5,6,7,8,9,10,11,12,13,14,15);
}

// B operand (32x16 f16): lane holds column n = tile*16 + (lane&15); K {kb..kb+7, kb+16..kb+23},
// kb = 32*kc + (lane>=16 ? 8 : 0). Weights stored [n][k] K-contiguous -> 2 b128 loads.
// Loop-invariant: hoisted into registers before the recurrence.
static __device__ __forceinline__ v16h ldsB(const _Float16* wl, int tile, int kc, int lane) {
  const int n = (tile << 4) + (lane & 15);
  const int k = (kc << 5) + ((lane >> 4) << 3);
  const v8h* p = (const v8h*)(wl + n * RS + k);
  const v8h lo = p[0];
  const v8h hi = p[2];
  return __builtin_shufflevector(lo, hi, 0,1,2,3,4,5,6,7,8,9,10,11,12,13,14,15);
}

#define WMMA_F16(a, b, c) \
  __builtin_amdgcn_wmma_f32_16x16x32_f16(false, (a), false, (b), (short)0, (c), false, false)

__global__ __launch_bounds__(128) void lstm_wmma_kernel(
    const float* __restrict__ x,     const float* __restrict__ W_ih,
    const float* __restrict__ W_hh,  const float* __restrict__ b_ih,
    const float* __restrict__ b_hh,  const float* __restrict__ W_out,
    const float* __restrict__ b_out, float* __restrict__ out)
{
  __shared__ _Float16 wlds[WROWS * RS];      // extended weights: K rows 0..49=W_hh, 50=W_ih, 51=bias
  __shared__ _Float16 hbuf[2][16 * RS];      // ping-pong A buffers: cols 0..49=h, 50=x_t, 51=1.0

  const int tid = threadIdx.x;

  // ---- zero LDS ----
  for (int i = tid; i < WROWS * RS; i += 128) wlds[i] = (_Float16)0.0f;
  {
    _Float16* hflat = &hbuf[0][0];
    for (int i = tid; i < 2 * 16 * RS; i += 128) hflat[i] = (_Float16)0.0f;
  }
  __syncthreads();

  // ---- fill weights (gate order i,f,g,o; N layout gate*64 + j, j<50 valid) ----
  for (int n = tid; n < WROWS; n += 128) {
    _Float16* row = &wlds[n * RS];
    if (n < 256) {
      const int g = n >> 6, j = n & 63;
      if (j < Hc) {
        const int rw = g * Hc + j;
        for (int k = 0; k < Hc; ++k) row[k] = (_Float16)W_hh[rw * Hc + k];
        row[50] = (_Float16)W_ih[rw];                       // I == 1
        row[51] = (_Float16)(b_ih[rw] + b_hh[rw]);
      }
    } else if (n == 256) {                                  // output N-tile, column 0
      for (int k = 0; k < Hc; ++k) row[k] = (_Float16)W_out[k];
      row[51] = (_Float16)b_out[0];                         // k=50 stays 0: x never leaks
    }
  }
  if (tid < 32) hbuf[tid >> 4][(tid & 15) * RS + 51] = (_Float16)1.0f;  // constant-1 column
  __syncthreads();

  const int w     = tid >> 5;                 // wave id: hidden group 0..3
  const int lane  = tid & 31;
  const int base  = blockIdx.x << 4;          // 16 batch rows per block
  const int rbase = (lane >> 4) << 3;         // C-tile M offset: 0 or 8

  // Branchless h-store column: pad lanes (j>=50) produce h == exactly 0, and
  // columns 52..63 are permanent zeros, so writing 0 there is harmless. Only
  // columns 50 (x) and 51 (const 1.0) must be protected: redirect them to 52/53.
  int col = (w << 4) + (lane & 15);
  if (col == 50 || col == 51) col += 2;       // one cndmask, computed once

  // ---- hoist all loop-invariant B operands into registers (10 x v16h = 80 VGPRs) ----
  const v16h Bi0 = ldsB(wlds, w,      0, lane);
  const v16h Bi1 = ldsB(wlds, w,      1, lane);
  const v16h Bf0 = ldsB(wlds, w + 4,  0, lane);
  const v16h Bf1 = ldsB(wlds, w + 4,  1, lane);
  const v16h Bg0 = ldsB(wlds, w + 8,  0, lane);
  const v16h Bg1 = ldsB(wlds, w + 8,  1, lane);
  const v16h Bo0 = ldsB(wlds, w + 12, 0, lane);
  const v16h Bo1 = ldsB(wlds, w + 12, 1, lane);
  const v16h Bq0 = ldsB(wlds, 16,     0, lane);   // output projection tile (used by wave 0)
  const v16h Bq1 = ldsB(wlds, 16,     1, lane);

  v8f cst = {};                               // cell state for this wave's 16 hidden units

  for (int t = 0; t < Tc; ++t) {
    _Float16* br = &hbuf[t & 1][0];           // holds h_{t-1}; gets x_t
    _Float16* bw = &hbuf[(t + 1) & 1][0];     // receives h_t

    if (w == 0 && lane < 16)
      br[lane * RS + 50] = (_Float16)x[(base + lane) * Tc + t];
    __syncthreads();                          // h_{t-1} + x_t visible; WAR on bw covered too

    const v16h a0 = ldsA(br, 0, lane);        // K 0..31
    const v16h a1 = ldsA(br, 1, lane);        // K 32..63 (incl. x, bias lane)

    // output projection for step t-1 (A holds h_{t-1}); wave-uniform branch -> EXEC all ones
    if (w == 0 && t > 0) {
      v8f z = {};
      v8f oc = WMMA_F16(a0, Bq0, z);
      oc = WMMA_F16(a1, Bq1, oc);
      if ((lane & 15) == 0) {
        #pragma unroll
        for (int r = 0; r < 8; ++r)
          out[(base + rbase + r) * Tc + (t - 1)] = oc[r];
      }
    }

    // gates for this wave's hidden group: tiles w, w+4, w+8, w+12 (i,f,g,o) — identical layouts
    v8f z = {};
    v8f ai = WMMA_F16(a0, Bi0, z);
    ai     = WMMA_F16(a1, Bi1, ai);
    v8f af = WMMA_F16(a0, Bf0, z);
    af     = WMMA_F16(a1, Bf1, af);
    v8f ag = WMMA_F16(a0, Bg0, z);
    ag     = WMMA_F16(a1, Bg1, ag);
    v8f ao = WMMA_F16(a0, Bo0, z);
    ao     = WMMA_F16(a1, Bo1, ao);

    _Float16* hs = bw + rbase * RS + col;     // unconditional stores, immediate offsets

    #pragma unroll
    for (int r = 0; r < 8; ++r) {
      const float ig = fsig(ai[r]);
      const float fg = fsig(af[r]);
      const float gg = ftanh(ag[r]);
      const float og = fsig(ao[r]);
      const float c  = fg * cst[r] + ig * gg;
      cst[r] = c;
      const float h  = og * ftanh(c);
      hs[r * RS] = (_Float16)h;
    }
  }

  // final output projection for t = Tc-1 (h_{Tc-1} sits in buffer Tc&1)
  __syncthreads();
  if (w == 0) {
    const _Float16* bf = &hbuf[Tc & 1][0];
    const v16h a0 = ldsA(bf, 0, lane);
    const v16h a1 = ldsA(bf, 1, lane);
    v8f z = {};
    v8f oc = WMMA_F16(a0, Bq0, z);
    oc = WMMA_F16(a1, Bq1, oc);
    if ((lane & 15) == 0) {
      #pragma unroll
      for (int r = 0; r < 8; ++r)
        out[(base + rbase + r) * Tc + (Tc - 1)] = oc[r];
    }
  }
}

extern "C" void kernel_launch(void* const* d_in, const int* in_sizes, int n_in,
                              void* d_out, int out_size, void* d_ws, size_t ws_size,
                              hipStream_t stream) {
  (void)in_sizes; (void)n_in; (void)d_ws; (void)ws_size; (void)out_size;
  const float* x     = (const float*)d_in[0];
  const float* W_ih  = (const float*)d_in[1];
  const float* W_hh  = (const float*)d_in[2];
  const float* b_ih  = (const float*)d_in[3];
  const float* b_hh  = (const float*)d_in[4];
  const float* W_out = (const float*)d_in[5];
  const float* b_out = (const float*)d_in[6];
  float* out = (float*)d_out;

  dim3 grid(Bc / 16);   // 256 blocks, one 16-row batch tile each
  dim3 block(128);      // 4 waves: hidden groups 0..3
  hipLaunchKernelGGL(lstm_wmma_kernel, grid, block, 0, stream,
                     x, W_ih, W_hh, b_ih, b_hh, W_out, b_out, out);
}
